// GraphormerMultiHeadAttention_88622355186287
// MI455X (gfx1250) — matrix operand order repaired and checked by reading, compiled, not run on hardware
//
#include <hip/hip_runtime.h>

typedef __attribute__((ext_vector_type(16))) _Float16 v16h;
typedef __attribute__((ext_vector_type(8)))  float    v8f;
typedef __attribute__((ext_vector_type(2)))  _Float16 h2;

#define N_SEQ 1024
#define DMODEL 512
#define NB 4
#define NH 8
#define HD 64
#define INF_DIST_F 1.0e9f

// ---------------- WMMA fragment loaders (CDNA5 ISA 7.12.2) -----
// A-fragment: 16x32 f16 (MxK). lane: row = lane&15, half = lane>>4.
// VGPR v holds K pair at kBase = half*16 + 2v.
__device__ __forceinline__ v16h load_a_f32(const float* base, int ld, int lane) {
  int r = lane & 15, hf = lane >> 4;
  const float* p = base + r * ld + hf * 16;
  v16h f;
#pragma unroll
  for (int v = 0; v < 8; ++v) {
    int kb = v * 2;
    float2 q = *(const float2*)(p + kb);
    f[2 * v] = (_Float16)q.x;
    f[2 * v + 1] = (_Float16)q.y;
  }
  return f;
}
__device__ __forceinline__ v16h load_a_f16(const _Float16* base, int ld, int lane) {
  int r = lane & 15, hf = lane >> 4;
  const _Float16* p = base + r * ld + hf * 16;
  v16h f;
#pragma unroll
  for (int v = 0; v < 8; ++v) {
    int kb = v * 2;
    h2 q = *(const h2*)(p + kb);
    f[2 * v] = q.x;
    f[2 * v + 1] = q.y;
  }
  return f;
}
// B-fragment: 32x16 f16 (KxN), loaded from B-transposed stored row-major [N][K].
// lane: col n = lane&15; VGPR v holds K pair at k = (lane>>4)*16 + 2v.
__device__ __forceinline__ v16h load_bt_f32(const float* base, int ld, int lane) {
  int n = lane & 15, hf = lane >> 4;
  const float* p = base + n * ld + hf * 16;
  v16h f;
#pragma unroll
  for (int v = 0; v < 8; ++v) {
    float2 q = *(const float2*)(p + 2 * v);
    f[2 * v] = (_Float16)q.x;
    f[2 * v + 1] = (_Float16)q.y;
  }
  return f;
}
__device__ __forceinline__ v16h load_bt_f16(const _Float16* base, int ld, int lane) {
  int n = lane & 15, hf = lane >> 4;
  const _Float16* p = base + n * ld + hf * 16;
  v16h f;
#pragma unroll
  for (int v = 0; v < 8; ++v) {
    h2 q = *(const h2*)(p + 2 * v);
    f[2 * v] = q.x;
    f[2 * v + 1] = q.y;
  }
  return f;
}

// ---------------- Kernel 1: per-batch max of reachable distances ----------
__global__ void kmaxdist(const float* __restrict__ dist, float* __restrict__ maxd) {
  __shared__ float red[256];
  int b = blockIdx.x;
  const float* p = dist + (size_t)b * N_SEQ * N_SEQ;
  float best = 0.0f;
  for (int i = threadIdx.x; i < N_SEQ * N_SEQ; i += blockDim.x) {
    float d = p[i];
    if (d != INF_DIST_F) best = fmaxf(best, d);
  }
  red[threadIdx.x] = best;
  __syncthreads();
  for (int s = 128; s > 0; s >>= 1) {
    if ((int)threadIdx.x < s) red[threadIdx.x] = fmaxf(red[threadIdx.x], red[threadIdx.x + s]);
    __syncthreads();
  }
  if (threadIdx.x == 0) maxd[b] = fmaxf(red[0], 1.0f);
}

// ---------------- Kernel 2: QKV projections (C = x @ W^T + b) -------------
// One wave per 16x64 output tile (4 WMMA accumulators share one A-fragment).
// Q,K -> [B,H,N,hd] f16 ; V -> [B,H,hd,N] f16 (transposed). A 64-col group
// lies inside exactly one head, so the head index is uniform per wave.
__global__ void kqkv(const float* __restrict__ x,
                     const float* __restrict__ Wq, const float* __restrict__ bq,
                     const float* __restrict__ Wk, const float* __restrict__ bk,
                     const float* __restrict__ Wv, const float* __restrict__ bv,
                     _Float16* __restrict__ Q, _Float16* __restrict__ K,
                     _Float16* __restrict__ Vt) {
  int lane = threadIdx.x & 31;
  int wid = threadIdx.x >> 5;
  int t = blockIdx.x * 8 + wid;          // 0..6143
  int mat = t >> 11;                     // 0=Q 1=K 2=V
  int rem = t & 2047;
  int row0 = (rem >> 3) << 4;            // 0..4080 over B*N
  int colg = (rem & 7) << 6;             // 0..448 over D (64-col group == head)
  const float* W = (mat == 0) ? Wq : (mat == 1) ? Wk : Wv;
  const float* bias = (mat == 0) ? bq : (mat == 1) ? bk : bv;

  v8f acc0 = {}, acc1 = {}, acc2 = {}, acc3 = {};
  for (int kk = 0; kk < DMODEL; kk += 32) {
    __builtin_prefetch(x + row0 * DMODEL + kk + 64, 0, 1);
    v16h a = load_a_f32(x + row0 * DMODEL + kk, DMODEL, lane);
    v16h b0 = load_bt_f32(W + (colg + 0 * 16) * DMODEL + kk, DMODEL, lane);
    v16h b1 = load_bt_f32(W + (colg + 1 * 16) * DMODEL + kk, DMODEL, lane);
    v16h b2 = load_bt_f32(W + (colg + 2 * 16) * DMODEL + kk, DMODEL, lane);
    v16h b3 = load_bt_f32(W + (colg + 3 * 16) * DMODEL + kk, DMODEL, lane);
    acc0 = __builtin_amdgcn_wmma_f32_16x16x32_f16(false, a, false, b0, (short)0, acc0, false, false);
    acc1 = __builtin_amdgcn_wmma_f32_16x16x32_f16(false, a, false, b1, (short)0, acc1, false, false);
    acc2 = __builtin_amdgcn_wmma_f32_16x16x32_f16(false, a, false, b2, (short)0, acc2, false, false);
    acc3 = __builtin_amdgcn_wmma_f32_16x16x32_f16(false, a, false, b3, (short)0, acc3, false, false);
  }

  int cl = lane & 15;
  int rowbase = (lane >> 4) * 8;
  int hh = colg >> 6;                    // head index (uniform)
  float bias0 = bias[colg + 0 * 16 + cl];
  float bias1 = bias[colg + 1 * 16 + cl];
  float bias2 = bias[colg + 2 * 16 + cl];
  float bias3 = bias[colg + 3 * 16 + cl];
#pragma unroll
  for (int r = 0; r < 8; ++r) {
    int m = row0 + rowbase + r;          // global row over B*N
    int b = m >> 10, n = m & 1023;
    _Float16 v0 = (_Float16)(acc0[r] + bias0);
    _Float16 v1 = (_Float16)(acc1[r] + bias1);
    _Float16 v2 = (_Float16)(acc2[r] + bias2);
    _Float16 v3 = (_Float16)(acc3[r] + bias3);
    if (mat == 2) {
      size_t hb = (size_t)(b * NH + hh) * HD;
      Vt[(hb + 0 * 16 + cl) * N_SEQ + n] = v0;
      Vt[(hb + 1 * 16 + cl) * N_SEQ + n] = v1;
      Vt[(hb + 2 * 16 + cl) * N_SEQ + n] = v2;
      Vt[(hb + 3 * 16 + cl) * N_SEQ + n] = v3;
    } else {
      _Float16* dst = (mat == 0) ? Q : K;
      size_t base = ((size_t)(b * NH + hh) * N_SEQ + n) * HD;
      dst[base + 0 * 16 + cl] = v0;
      dst[base + 1 * 16 + cl] = v1;
      dst[base + 2 * 16 + cl] = v2;
      dst[base + 3 * 16 + cl] = v3;
    }
  }
}

// ---------------- Kernel 3: attention, one wave per (b,h,16-row block) ----
// LDS: 16x1024 f32 score strip (64KB) + 16x1024 f16 P strip (32KB) = 96KB
__global__ void kattn(const float* __restrict__ dist, const float* __restrict__ maxd,
                      const _Float16* __restrict__ Q, const _Float16* __restrict__ K,
                      const _Float16* __restrict__ Vt, _Float16* __restrict__ AO) {
  extern __shared__ char smem[];
  float* sc = (float*)smem;                                  // [16][1024] f32
  _Float16* pm = (_Float16*)(smem + 16 * N_SEQ * 4);         // [16][1024] f16

  int lane = threadIdx.x & 31;
  int t = blockIdx.x;                // 0..2047
  int b = t >> 9;
  int rem = t & 511;
  int h = rem >> 6;
  int i0 = (rem & 63) << 4;

  const _Float16* Qh = Q + (size_t)(b * NH + h) * N_SEQ * HD;
  const _Float16* Kh = K + (size_t)(b * NH + h) * N_SEQ * HD;
  const _Float16* Vh = Vt + (size_t)(b * NH + h) * HD * N_SEQ;
  const float* db = dist + ((size_t)b << 20);
  float inv_maxd = 1.0f / maxd[b];

  int col = lane & 15;
  int rowbase = (lane >> 4) * 8;

  // ---- phase 1: scores = QK^T/8 - 10*dist/maxd, masked -> LDS strip ----
  // Q strip (16x64) is loop-invariant: hoist its two A-fragments.
  v16h qa0 = load_a_f16(Qh + i0 * HD + 0, HD, lane);
  v16h qa1 = load_a_f16(Qh + i0 * HD + 32, HD, lane);
  for (int j0 = 0; j0 < N_SEQ; j0 += 16) {
    v8f acc = {};
    v16h bt0 = load_bt_f16(Kh + j0 * HD + 0, HD, lane);   // B^T == K row-major
    v16h bt1 = load_bt_f16(Kh + j0 * HD + 32, HD, lane);
    acc = __builtin_amdgcn_wmma_f32_16x16x32_f16(false, qa0, false, bt0, (short)0, acc, false, false);
    acc = __builtin_amdgcn_wmma_f32_16x16x32_f16(false, qa1, false, bt1, (short)0, acc, false, false);
    int j = j0 + col;
#pragma unroll
    for (int r = 0; r < 8; ++r) {
      int il = rowbase + r;
      int i = i0 + il;
      float dd = db[((size_t)i << 10) + j];
      float s = (dd == INF_DIST_F) ? -__builtin_inff()
                                   : acc[r] * 0.125f - 10.0f * dd * inv_maxd;
      sc[il * N_SEQ + j] = s;
    }
  }
  __syncthreads();

  // ---- phase 2: wave-cooperative softmax (unnormalized exp into pm) ----
  float rinv[8];
#pragma unroll
  for (int r = 0; r < 16; ++r) {
    float m = -__builtin_inff();
    for (int c = lane; c < N_SEQ; c += 32) m = fmaxf(m, sc[r * N_SEQ + c]);
#pragma unroll
    for (int s = 16; s > 0; s >>= 1) m = fmaxf(m, __shfl_xor(m, s, 32));
    if (m == -__builtin_inff()) m = 0.0f;
    float sum = 0.0f;
    for (int c = lane; c < N_SEQ; c += 32) {
      float e = __expf(sc[r * N_SEQ + c] - m);
      pm[r * N_SEQ + c] = (_Float16)e;
      sum += e;
    }
#pragma unroll
    for (int s = 16; s > 0; s >>= 1) sum += __shfl_xor(sum, s, 32);
    if ((r >> 3) == (lane >> 4)) rinv[r & 7] = 1.0f / sum;
  }
  __syncthreads();

  // ---- phase 3: out(16x64) = P(16x1024) @ V(1024x64), then normalize ----
  v8f acc0 = {}, acc1 = {}, acc2 = {}, acc3 = {};
  for (int j0 = 0; j0 < N_SEQ; j0 += 32) {
    v16h a = load_a_f16(pm + j0, N_SEQ, lane);
    v16h b0 = load_bt_f16(Vh + 0 * 16 * N_SEQ + j0, N_SEQ, lane);
    v16h b1 = load_bt_f16(Vh + 1 * 16 * N_SEQ + j0, N_SEQ, lane);
    v16h b2 = load_bt_f16(Vh + 2 * 16 * N_SEQ + j0, N_SEQ, lane);
    v16h b3 = load_bt_f16(Vh + 3 * 16 * N_SEQ + j0, N_SEQ, lane);
    acc0 = __builtin_amdgcn_wmma_f32_16x16x32_f16(false, a, false, b0, (short)0, acc0, false, false);
    acc1 = __builtin_amdgcn_wmma_f32_16x16x32_f16(false, a, false, b1, (short)0, acc1, false, false);
    acc2 = __builtin_amdgcn_wmma_f32_16x16x32_f16(false, a, false, b2, (short)0, acc2, false, false);
    acc3 = __builtin_amdgcn_wmma_f32_16x16x32_f16(false, a, false, b3, (short)0, acc3, false, false);
  }
#pragma unroll
  for (int r = 0; r < 8; ++r) {
    int il = rowbase + r;
    int n = i0 + il;
    float s = rinv[r];
    size_t base = ((size_t)(b * N_SEQ + n)) * DMODEL + h * HD;
    AO[base + 0 * 16 + col] = (_Float16)(acc0[r] * s);
    AO[base + 1 * 16 + col] = (_Float16)(acc1[r] * s);
    AO[base + 2 * 16 + col] = (_Float16)(acc2[r] * s);
    AO[base + 3 * 16 + col] = (_Float16)(acc3[r] * s);
  }
}

// ---------------- Kernel 4: output projection (f32 result) ---------------
// One wave per 16x64 tile: 4 accumulators share one A-fragment per k-step.
__global__ void kproj(const _Float16* __restrict__ AO, const float* __restrict__ Wo,
                      const float* __restrict__ bo, float* __restrict__ out) {
  int lane = threadIdx.x & 31;
  int wid = threadIdx.x >> 5;
  int t = blockIdx.x * 8 + wid;        // 0..2047
  int row0 = (t >> 3) << 4;            // 0..4080
  int colg = (t & 7) << 6;             // 0..448

  v8f acc0 = {}, acc1 = {}, acc2 = {}, acc3 = {};
  for (int kk = 0; kk < DMODEL; kk += 32) {
    __builtin_prefetch(AO + row0 * DMODEL + kk + 64, 0, 1);
    v16h a = load_a_f16(AO + row0 * DMODEL + kk, DMODEL, lane);
    v16h b0 = load_bt_f32(Wo + (colg + 0 * 16) * DMODEL + kk, DMODEL, lane);
    v16h b1 = load_bt_f32(Wo + (colg + 1 * 16) * DMODEL + kk, DMODEL, lane);
    v16h b2 = load_bt_f32(Wo + (colg + 2 * 16) * DMODEL + kk, DMODEL, lane);
    v16h b3 = load_bt_f32(Wo + (colg + 3 * 16) * DMODEL + kk, DMODEL, lane);
    acc0 = __builtin_amdgcn_wmma_f32_16x16x32_f16(false, a, false, b0, (short)0, acc0, false, false);
    acc1 = __builtin_amdgcn_wmma_f32_16x16x32_f16(false, a, false, b1, (short)0, acc1, false, false);
    acc2 = __builtin_amdgcn_wmma_f32_16x16x32_f16(false, a, false, b2, (short)0, acc2, false, false);
    acc3 = __builtin_amdgcn_wmma_f32_16x16x32_f16(false, a, false, b3, (short)0, acc3, false, false);
  }
  int cl = lane & 15;
  float bias0 = bo[colg + 0 * 16 + cl];
  float bias1 = bo[colg + 1 * 16 + cl];
  float bias2 = bo[colg + 2 * 16 + cl];
  float bias3 = bo[colg + 3 * 16 + cl];
  int rowbase = (lane >> 4) * 8;
#pragma unroll
  for (int r = 0; r < 8; ++r) {
    size_t base = (size_t)(row0 + rowbase + r) * DMODEL + colg;
    out[base + 0 * 16 + cl] = acc0[r] + bias0;
    out[base + 1 * 16 + cl] = acc1[r] + bias1;
    out[base + 2 * 16 + cl] = acc2[r] + bias2;
    out[base + 3 * 16 + cl] = acc3[r] + bias3;
  }
}

// ---------------- launch ---------------------------------------------------
extern "C" void kernel_launch(void* const* d_in, const int* in_sizes, int n_in,
                              void* d_out, int out_size, void* d_ws, size_t ws_size,
                              hipStream_t stream) {
  const float* x = (const float*)d_in[0];
  const float* dist = (const float*)d_in[1];
  const float* Wq = (const float*)d_in[2];
  const float* bq = (const float*)d_in[3];
  const float* Wk = (const float*)d_in[4];
  const float* bk = (const float*)d_in[5];
  const float* Wv = (const float*)d_in[6];
  const float* bv = (const float*)d_in[7];
  const float* Wo = (const float*)d_in[8];
  const float* bo = (const float*)d_in[9];
  float* out = (float*)d_out;

  const size_t QKV_ELEMS = (size_t)NB * NH * N_SEQ * HD;  // 2,097,152 halves each
  char* ws = (char*)d_ws;
  float* maxd = (float*)ws;                       // 4 floats
  _Float16* Q = (_Float16*)(ws + 256);
  _Float16* K = Q + QKV_ELEMS;
  _Float16* Vt = K + QKV_ELEMS;
  _Float16* AO = Vt + QKV_ELEMS;                  // [B,N,D] f16

  kmaxdist<<<NB, 256, 0, stream>>>(dist, maxd);
  // 3 matrices * 256 row-tiles * 8 col-groups = 6144 waves / 8 per block
  kqkv<<<768, 256, 0, stream>>>(x, Wq, bq, Wk, bk, Wv, bv, Q, K, Vt);
  // one wave (workgroup) per (b, h, 16-row block); 96KB dynamic LDS
  kattn<<<NB * NH * (N_SEQ / 16), 32, 16 * N_SEQ * 4 + 16 * N_SEQ * 2, stream>>>(
      dist, maxd, Q, K, Vt, AO);
  // 256 row-tiles * 8 col-groups = 2048 waves / 8 per block
  kproj<<<256, 256, 0, stream>>>(AO, Wo, bo, out);
}
